// GDGCN_63771674411187
// MI455X (gfx1250) — compile-verified, matrix operand order checked by reading, not run on hardware
//
#include <hip/hip_runtime.h>
#include <hip/hip_bf16.h>

// ---------------------------------------------------------------------------
// GDGCN fused dynamic-graph conv for gfx1250 (MI455X), wave32 + WMMA f16.
//
//   N=8192, D=32, B*C=16, T=12, J=B*C*T=192
//   core = tv @ k ; P = nodevec1 @ core                        (prep)
//   logit[n,m] = P[n,:] . V2[m,:]   (v_wmma_f32_16x16x32_f16, K = D = 32)
//   s[n] = sum_m exp(relu(logit))                              (rowsum)
//   out[m,j] = sum_n exp(relu(logit[n,m]))/s[n] * XhT[j,n]     (aggregate)
//
// The 268MB adjacency is never materialized: logit tiles are recomputed.
// All persistent operands (P, V2, XhT: ~4MB f16) are L2-resident; the
// aggregate kernel stages each 32-n chunk of XhT/P into LDS ONCE per
// workgroup via async global->LDS DMA (double buffered, ASYNCcnt), so the
// 4 waves share it instead of each streaming it from L2.
//
// f16 range: softmax weights ~1/N, so fold SCALE_E=256 into 1/s and divide
// back at the final store.
// ---------------------------------------------------------------------------

typedef __attribute__((ext_vector_type(16))) _Float16 v16h;
typedef __attribute__((ext_vector_type(8)))  float    v8f;
typedef __attribute__((ext_vector_type(4)))  int      v4i;

#define NN     8192
#define DD     32
#define BC     16
#define TT     12
#define JJ     192          // BC*TT
#define NTILES 512          // NN/16
#define NCHUNK 32
#define NITERS (NN / NCHUNK)
#define SCALE_E 256.0f
#define INV_SCALE_E 0.00390625f

// ---- async global->LDS (gfx1250) with graceful fallback -------------------
#if defined(__has_builtin)
#if __has_builtin(__builtin_amdgcn_global_load_async_to_lds_b128)
#define HAVE_ASYNC 1
#endif
#endif
#ifndef HAVE_ASYNC
#define HAVE_ASYNC 0
#endif

#if HAVE_ASYNC
typedef __attribute__((address_space(1))) v4i g_v4i;   // global int4
typedef __attribute__((address_space(3))) v4i l_v4i;   // LDS int4
__device__ __forceinline__ void async_b128(const void* g, void* l) {
  // (global v4i*, lds v4i*, imm offset, cpol); AS casts via integers
  // (AS3 pointer = low 32 bits of the generic LDS address).
  __builtin_amdgcn_global_load_async_to_lds_b128(
      (g_v4i*)(unsigned long long)(uintptr_t)g,
      (l_v4i*)(unsigned int)(uintptr_t)l, 0, 0);
}
#if __has_builtin(__builtin_amdgcn_s_wait_asynccnt)
#define WAIT_ASYNC(n) __builtin_amdgcn_s_wait_asynccnt(n)
#else
#define WAIT_ASYNC(n) asm volatile("s_wait_asynccnt %0" ::"n"(n) : "memory")
#endif
#endif

// ---- WMMA wrapper ---------------------------------------------------------
__device__ __forceinline__ v8f wmma_f16(v16h a, v16h b, v8f c) {
  return __builtin_amdgcn_wmma_f32_16x16x32_f16(false, a, false, b,
                                                (short)0, c, false, false);
}

// ---- A/B fragment loader for a row-major 16x32 f16 tile (global) ----------
// ISA 7.12.2: lane L -> row M = L%16; per lane two 16B loads at byte
// offsets {hi*16} and {32 + hi*16} within the 64B row.
__device__ __forceinline__ v16h load_frag32(const _Float16* __restrict__ base,
                                            int stride_h, int lane) {
  const int r  = lane & 15;
  const int hi = (lane >> 4) & 1;
  const _Float16* p = base + (size_t)r * stride_h + hi * 8;
  union { v16h h; uint4 q[2]; } f;
  f.q[0] = *reinterpret_cast<const uint4*>(p);
  f.q[1] = *reinterpret_cast<const uint4*>(p + 16);
  return f.h;
}

// ===========================================================================
// Kernel A: core, P -> f16, V2 -> f16, x -> XhT[j][n] f16.  grid 32 x 256.
// ===========================================================================
__global__ void __launch_bounds__(256)
gdgcn_prep(const float* __restrict__ x,
           const float* __restrict__ nv1,
           const float* __restrict__ nv2,
           const float* __restrict__ timevec,
           const float* __restrict__ kk,
           const int*   __restrict__ time_ind,
           _Float16* __restrict__ Ph,
           _Float16* __restrict__ V2h,
           _Float16* __restrict__ XhT) {
  __shared__ float core[DD][DD + 1];
  const int tid = threadIdx.x;
  const float* tv = timevec + (size_t)time_ind[0] * DD;

  for (int idx = tid; idx < DD * DD; idx += 256) {
    const int e = idx >> 5, f = idx & 31;
    float s = 0.f;
#pragma unroll
    for (int d = 0; d < DD; ++d) s += tv[d] * kk[d * (DD * DD) + e * DD + f];
    core[e][f] = s;
  }
  __syncthreads();

  const int n = blockIdx.x * 256 + tid;

  float a[DD];
#pragma unroll
  for (int e = 0; e < DD; ++e) a[e] = nv1[(size_t)n * DD + e];
#pragma unroll
  for (int f = 0; f < DD; ++f) {
    float s = 0.f;
#pragma unroll
    for (int e = 0; e < DD; ++e) s += a[e] * core[e][f];
    Ph [(size_t)n * DD + f] = (_Float16)s;
    V2h[(size_t)n * DD + f] = (_Float16)nv2[(size_t)n * DD + f];
  }

#pragma unroll
  for (int bc = 0; bc < BC; ++bc) {
    const float* xp = x + (size_t)bc * (NN * TT) + (size_t)n * TT;
#pragma unroll
    for (int t = 0; t < TT; ++t)
      XhT[(size_t)(bc * TT + t) * NN + n] = (_Float16)xp[t];
  }
}

// ===========================================================================
// Kernel B: s[n] = sum_m exp(relu(logit[n,m])).  One wave per PAIR of
// 16-row n-tiles so each V2 B-fragment load feeds two WMMAs.  grid 64 x 128.
// ===========================================================================
__global__ void __launch_bounds__(128)
gdgcn_rowsum(const _Float16* __restrict__ Ph,
             const _Float16* __restrict__ V2h,
             float* __restrict__ sinv) {
  const int wave = threadIdx.x >> 5;
  const int lane = threadIdx.x & 31;
  const int pair = blockIdx.x * 4 + wave;        // 0..255 -> rows pair*32..+31

  const v16h a0 = load_frag32(Ph + (size_t)(pair * 32)      * DD, DD, lane);
  const v16h a1 = load_frag32(Ph + (size_t)(pair * 32 + 16) * DD, DD, lane);

  float acc0[8], acc1[8];
#pragma unroll
  for (int v = 0; v < 8; ++v) { acc0[v] = 0.f; acc1[v] = 0.f; }

  for (int mt = 0; mt < NTILES; ++mt) {
    const v16h b = load_frag32(V2h + (size_t)mt * 16 * DD, DD, lane);
    v8f c0 = {}, c1 = {};
    c0 = wmma_f16(a0, b, c0);
    c1 = wmma_f16(a1, b, c1);
#pragma unroll
    for (int v = 0; v < 8; ++v) {
      acc0[v] += __expf(fmaxf(c0[v], 0.f));
      acc1[v] += __expf(fmaxf(c1[v], 0.f));
    }
  }

  // D layout: lane holds col N=lane%16, row M = v + 8*(lane/16):
  // rows v in lanes 0..15, rows v+8 in lanes 16..31 -> 16-wide reductions.
#pragma unroll
  for (int v = 0; v < 8; ++v) {
#pragma unroll
    for (int m = 8; m >= 1; m >>= 1) {
      acc0[v] += __shfl_xor(acc0[v], m, 16);
      acc1[v] += __shfl_xor(acc1[v], m, 16);
    }
  }
  const int nb = pair * 32;
  if (lane == 0) {
#pragma unroll
    for (int v = 0; v < 8; ++v) {
      sinv[nb + v]      = SCALE_E / acc0[v];
      sinv[nb + 16 + v] = SCALE_E / acc1[v];
    }
  }
  if (lane == 16) {
#pragma unroll
    for (int v = 0; v < 8; ++v) {
      sinv[nb + 8 + v]  = SCALE_E / acc0[v];
      sinv[nb + 24 + v] = SCALE_E / acc1[v];
    }
  }
}

// ===========================================================================
// Kernel C: out[m,j] = sum_n (exp(relu(logit[n,m])) * sinv[n]) * XhT[j,n].
// 4 waves/WG, one 16-wide m-tile per wave, 12 persistent f32 acc fragments.
// Per 32-n chunk the WG stages XhT (12KB) + Ph (2KB) into LDS once (async
// DMA, double buffered) and all waves consume it.  grid 128 x 128.
// ===========================================================================
__global__ void __launch_bounds__(128)
gdgcn_aggregate(const _Float16* __restrict__ Ph,
                const _Float16* __restrict__ V2h,
                const _Float16* __restrict__ XhT,
                const float*    __restrict__ sinv,
                float* __restrict__ out) {
  // Row stride 48 halves (96B): keeps every b128 LDS access 16B aligned and
  // spreads banks.  Xs: 2 x 192 x 48, Ps: 2 x 32 x 48, Elds: 4 x 16 x 48.
  __shared__ __align__(16) _Float16 Xs[2][JJ][48];
  __shared__ __align__(16) _Float16 Ps[2][NCHUNK][48];
  __shared__ __align__(16) _Float16 Elds[4][16][48];

  const int tid   = threadIdx.x;
  const int wave  = tid >> 5;
  const int lane  = tid & 31;
  const int r     = lane & 15;
  const int hi    = (lane >> 4) & 1;
  const int mbase = (blockIdx.x * 4 + wave) * 16;

  // Staging partition: 128 threads, 4 threads per 64B row segment.
  const int srow = tid >> 2;   // 0..31
  const int sseg = tid & 3;    // 0..3 (16B each)

  // 7 staging ops per chunk per wave: 6 x (32 rows of XhT) + 1 x (32 rows P).
#if HAVE_ASYNC
#define STAGE_CHUNK(buf, n0base)                                              \
  do {                                                                        \
    _Pragma("unroll")                                                         \
    for (int q = 0; q < 6; ++q) {                                             \
      const int j = q * 32 + srow;                                            \
      async_b128(XhT + (size_t)j * NN + (n0base) + sseg * 8,                  \
                 &Xs[(buf)][j][sseg * 8]);                                    \
    }                                                                         \
    async_b128(Ph + (size_t)((n0base) + srow) * DD + sseg * 8,                \
               &Ps[(buf)][srow][sseg * 8]);                                   \
  } while (0)
#else
#define STAGE_CHUNK(buf, n0base)                                              \
  do {                                                                        \
    _Pragma("unroll")                                                         \
    for (int q = 0; q < 6; ++q) {                                             \
      const int j = q * 32 + srow;                                            \
      *reinterpret_cast<uint4*>(&Xs[(buf)][j][sseg * 8]) =                    \
          *reinterpret_cast<const uint4*>(XhT + (size_t)j * NN + (n0base) +   \
                                          sseg * 8);                          \
    }                                                                         \
    *reinterpret_cast<uint4*>(&Ps[(buf)][srow][sseg * 8]) =                   \
        *reinterpret_cast<const uint4*>(Ph + (size_t)((n0base) + srow) * DD + \
                                        sseg * 8);                            \
  } while (0)
#endif

  // Loop-invariant logit B-fragment: V2 rows of this wave's m-tile.
  const v16h bv = load_frag32(V2h + (size_t)mbase * DD, DD, lane);

  v8f acc[12];
#pragma unroll
  for (int jt = 0; jt < 12; ++jt) acc[jt] = (v8f){};

  STAGE_CHUNK(0, 0);

  for (int it = 0; it < NITERS; ++it) {
    const int cur = it & 1;
    const int n0  = it * NCHUNK;
    const bool more = (it + 1 < NITERS);
    if (more) STAGE_CHUNK(cur ^ 1, n0 + NCHUNK);   // prefetch next chunk
#if HAVE_ASYNC
    if (more) WAIT_ASYNC(7); else WAIT_ASYNC(0);   // chunk `it` landed
#endif
    __syncthreads();                               // cross-wave visibility

    // --- logits -> exp(relu)*sinv -> f16, transposed into private E slice --
#pragma unroll
    for (int sub = 0; sub < 2; ++sub) {
      union { v16h h; uint4 q[2]; } av;
      av.q[0] = *reinterpret_cast<const uint4*>(&Ps[cur][sub * 16 + r][hi * 8]);
      av.q[1] = *reinterpret_cast<const uint4*>(&Ps[cur][sub * 16 + r][hi * 8 + 16]);
      v8f d = {};
      d = wmma_f16(av.h, bv, d);

      const int nb = n0 + sub * 16 + hi * 8;       // D row (=n_local)=v+8*hi
      union { _Float16 h[8]; uint4 q; } pk;
#pragma unroll
      for (int v = 0; v < 8; ++v)
        pk.h[v] = (_Float16)(__expf(fmaxf(d[v], 0.f)) * sinv[nb + v]);
      *reinterpret_cast<uint4*>(&Elds[wave][r][sub * 16 + hi * 8]) = pk.q;
    }

    // --- E^T tile back as A-fragment (same-wave DScnt RAW; no barrier) -----
    union { v16h h; uint4 q[2]; } ae;
    ae.q[0] = *reinterpret_cast<const uint4*>(&Elds[wave][r][hi * 8]);
    ae.q[1] = *reinterpret_cast<const uint4*>(&Elds[wave][r][hi * 8 + 16]);

    // --- 12 aggregation WMMAs, B-fragments from the shared X stage ---------
#pragma unroll
    for (int jt = 0; jt < 12; ++jt) {
      union { v16h h; uint4 q[2]; } bx;
      bx.q[0] = *reinterpret_cast<const uint4*>(&Xs[cur][jt * 16 + r][hi * 8]);
      bx.q[1] = *reinterpret_cast<const uint4*>(&Xs[cur][jt * 16 + r][hi * 8 + 16]);
      acc[jt] = wmma_f16(ae.h, bx.h, acc[jt]);
    }

    __syncthreads();   // all reads of buf `cur` done before it is re-staged
  }

  // Store: D: m_local = v + 8*hi, j_local = r;  j = bc*12 + t.
#pragma unroll
  for (int jt = 0; jt < 12; ++jt) {
#pragma unroll
    for (int v = 0; v < 8; ++v) {
      const int m  = mbase + v + 8 * hi;
      const int j  = jt * 16 + r;
      const int bc = j / TT;
      const int t  = j - bc * TT;
      out[(size_t)bc * (NN * TT) + (size_t)m * TT + t] = acc[jt][v] * INV_SCALE_E;
    }
  }
}

// ===========================================================================
// Host-side launcher
// ===========================================================================
extern "C" void kernel_launch(void* const* d_in, const int* in_sizes, int n_in,
                              void* d_out, int out_size, void* d_ws, size_t ws_size,
                              hipStream_t stream) {
  const float* x        = (const float*)d_in[0];
  const float* nodevec1 = (const float*)d_in[1];
  const float* nodevec2 = (const float*)d_in[2];
  const float* timevec  = (const float*)d_in[3];
  const float* kk       = (const float*)d_in[4];
  const int*   time_ind = (const int*)  d_in[5];
  float*       out      = (float*)d_out;

  // Workspace: Ph 512KB | V2h 512KB | XhT 3MB | sinv 32KB  (~4.1MB total)
  char* ws = (char*)d_ws;
  _Float16* Ph   = (_Float16*)(ws);
  _Float16* V2h  = (_Float16*)(ws + (size_t)NN * DD * 2);
  _Float16* XhT  = (_Float16*)(ws + (size_t)NN * DD * 4);
  float*    sinv = (float*)   (ws + (size_t)NN * DD * 4 + (size_t)JJ * NN * 2);

  gdgcn_prep<<<dim3(NN / 256), dim3(256), 0, stream>>>(
      x, nodevec1, nodevec2, timevec, kk, time_ind, Ph, V2h, XhT);

  gdgcn_rowsum<<<dim3(NTILES / 8), dim3(128), 0, stream>>>(Ph, V2h, sinv);

  gdgcn_aggregate<<<dim3(NN / 64), dim3(128), 0, stream>>>(
      Ph, V2h, XhT, sinv, out);
}
